// PointNetPolylineEncoder_18511309045816
// MI455X (gfx1250) — matrix operand order repaired and checked by reading, not compiled, not run on hardware
//
#include <hip/hip_runtime.h>
#include <hip/hip_bf16.h>

typedef _Float16 v16h __attribute__((ext_vector_type(16)));
typedef _Float16 v8h  __attribute__((ext_vector_type(8)));
typedef float    v8f  __attribute__((ext_vector_type(8)));

#define EPS 1e-5f

// Problem constants
#define Bn 16
#define Pn 768
#define Nn 32
#define Cn 9
#define Hn 128
#define On 256

// f16 weight image layout in d_ws (element offsets, halves)
#define OFF_PRE 0            // [128][32]  (C padded 9->32)
#define OFF_W1  4096         // [128][256]
#define OFF_W2  36864        // [128][128]
#define NHALF   53248        // total halves
// float region (BN folded) starts right after halves
// fs: spre[128] tpre[128] s1[128] t1[128] s2[128] t2[128]

__device__ __forceinline__ v8f wmma16(v16h a, v16h b, v8f c) {
    return __builtin_amdgcn_wmma_f32_16x16x32_f16(false, a, false, b, (short)0, c, false, false);
}

// One K=32 step: 1 A fragment (from LDS), 4 B fragments (f16 weights, L2), 4 WMMAs.
__device__ __forceinline__ void do_gemm(const _Float16* __restrict__ A, int sA, int mrow,
                                        const _Float16* __restrict__ Bw, int sB, int nrow0,
                                        int kbA, int kbB, int chunkoff, int khalf,
                                        v8f acc[4]) {
    // A layout (16x32 f16): lane<16 -> K base 0, lane>=16 -> K base 16; halves 0..15 = K+0..15
    // => two adjacent 16B chunks.
    const _Float16* ap = A + mrow * sA + kbA + chunkoff;
    v8h a0 = *(const v8h*)(ap);
    v8h a1 = *(const v8h*)(ap + 8);
    v16h av = __builtin_shufflevector(a0, a1, 0,1,2,3,4,5,6,7,8,9,10,11,12,13,14,15);
#pragma unroll
    for (int t = 0; t < 4; ++t) {
        // B layout (32x16 f16): lane = N row, 16 contiguous K halves per lane (32B load)
        const v16h bv = *(const v16h*)(Bw + (nrow0 + 16 * t) * sB + kbB + khalf);
        acc[t] = wmma16(av, bv, acc[t]);
    }
}

// Epilogue: y = relu(acc*scale + shift) * rowmask  -> f16 LDS
__device__ __forceinline__ void epilogue(v8f acc[4], int nt0, int lane, int mtile,
                                         const float* __restrict__ sc, const float* __restrict__ sh,
                                         const float rowm[8], _Float16* __restrict__ dst, int sD) {
    int r = lane & 15, hi = lane >> 4;
#pragma unroll
    for (int t = 0; t < 4; ++t) {
        int feat = (nt0 + t) * 16 + r;
        float s = sc[feat], b = sh[feat];
#pragma unroll
        for (int v = 0; v < 8; ++v) {
            int m = mtile * 16 + v + 8 * hi;   // C/D layout: VGPR v -> M = v (+8 for hi half)
            float y = fmaxf(acc[t][v] * s + b, 0.0f) * rowm[v];
            dst[m * sD + feat] = (_Float16)y;
        }
    }
}

// Prolog: convert/pad weights to f16, fold BN into (scale, shift).
__global__ __launch_bounds__(256)
void prep_kernel(const float* __restrict__ Wpre, const float* __restrict__ W1,
                 const float* __restrict__ W2,
                 const float* gp, const float* bp, const float* mp, const float* vp,
                 const float* g1, const float* b1, const float* m1, const float* v1,
                 const float* g2, const float* b2, const float* m2, const float* v2,
                 _Float16* __restrict__ w16, float* __restrict__ fs) {
    int gid = blockIdx.x * 256 + threadIdx.x;
    if (gid < 4096) {                         // W_pre padded [128][32]
        int o = gid >> 5, k = gid & 31;
        w16[gid] = (_Float16)(k < Cn ? Wpre[o * Cn + k] : 0.0f);
    } else if (gid < 36864) {                 // W1 [128][256]
        w16[gid] = (_Float16)W1[gid - 4096];
    } else if (gid < NHALF) {                 // W2 [128][128]
        w16[gid] = (_Float16)W2[gid - 36864];
    }
    if (gid < 128) {
        float s = gp[gid] * rsqrtf(vp[gid] + EPS);
        fs[gid] = s; fs[128 + gid] = bp[gid] - mp[gid] * s;
    } else if (gid < 256) {
        int i = gid - 128; float s = g1[i] * rsqrtf(v1[i] + EPS);
        fs[256 + i] = s; fs[384 + i] = b1[i] - m1[i] * s;
    } else if (gid < 384) {
        int i = gid - 256; float s = g2[i] * rsqrtf(v2[i] + EPS);
        fs[512 + i] = s; fs[640 + i] = b2[i] - m2[i] * s;
    }
}

// Fused encoder: one workgroup = 2 polylines = 64 point-rows; all activations in LDS.
__global__ __launch_bounds__(256)
void encoder_kernel(const float* __restrict__ x, const unsigned char* __restrict__ mask,
                    const _Float16* __restrict__ w16, const float* __restrict__ fs,
                    const float* __restrict__ Wo1, const float* __restrict__ bo1,
                    const float* __restrict__ Wo2, const float* __restrict__ bo2,
                    float* __restrict__ out) {
    __shared__ _Float16 s_x[64 * 32];        // padded input, f16
    __shared__ _Float16 s_f[64 * 128];       // layer-pre out / reused for layer2 out
    __shared__ _Float16 s_h[64 * 128];       // layer1 out
    __shared__ _Float16 s_pool16[2 * 128];   // pooled f (f16) -> second half of concat
    __shared__ float    s_maskf[64];
    __shared__ float    s_pool2[2 * 128];
    __shared__ float    s_o1[2 * 128];
    __shared__ float    s_valid[2];

    const int tid = threadIdx.x;
    const int lane = tid & 31, wave = tid >> 5;
    const int mtile = wave & 3;              // 4 M-tiles of 16 rows
    const int nt0 = (wave >> 2) * 4;         // waves 0-3: N-tiles 0-3; waves 4-7: 4-7
    const int r15 = lane & 15, hi = lane >> 4;
    const int mrow = mtile * 16 + r15;
    const int nrow0 = nt0 * 16 + r15;
    const int chunkoff = 16 * hi, khalf = 16 * hi;

    const _Float16* Wpre16 = w16 + OFF_PRE;
    const _Float16* W1_16  = w16 + OFF_W1;
    const _Float16* W2_16  = w16 + OFF_W2;

    const long rowbase = (long)blockIdx.x * 64;      // global point-row base

    // ---- Stage 1: stage input tile (f32 -> f16, pad K to 32) + mask ----
#pragma unroll
    for (int i = tid; i < 64 * 32; i += 256) s_x[i] = (_Float16)0.0f;
    __syncthreads();
    for (int i = tid; i < 64 * Cn; i += 256) {
        int rrow = i / Cn, c = i - rrow * Cn;
        s_x[rrow * 32 + c] = (_Float16)x[(rowbase + rrow) * Cn + c];
    }
    if (tid < 64) s_maskf[tid] = mask[rowbase + tid] ? 1.0f : 0.0f;
    __syncthreads();

    // ---- Stage 2: layer pre (K=32, one WMMA step per tile) ----
    {
        v8f acc[4] = {};
        do_gemm(s_x, 32, mrow, Wpre16, 32, nrow0, 0, 0, chunkoff, khalf, acc);
        float mm[8];
#pragma unroll
        for (int v = 0; v < 8; ++v) mm[v] = s_maskf[mtile * 16 + v + 8 * hi];
        epilogue(acc, nt0, lane, mtile, fs + 0, fs + 128, mm, s_f, 128);
    }
    __syncthreads();

    // ---- Stage 3: max-pool over 32 points -> pooled (f16 broadcast source) ----
    {
        int poly = tid >> 7, feat = tid & 127;
        float mx = 0.0f;                       // all values >= 0 (relu + zero-mask)
#pragma unroll 4
        for (int rr = 0; rr < Nn; ++rr)
            mx = fmaxf(mx, (float)s_f[(poly * Nn + rr) * 128 + feat]);
        s_pool16[poly * 128 + feat] = (_Float16)mx;
    }
    __syncthreads();

    // ---- Stage 4: layer 1, K=256 = [f (k<128) | pooled (k>=128)] ----
    {
        v8f acc[4] = {};
#pragma unroll
        for (int kt = 0; kt < 4; ++kt)
            do_gemm(s_f, 128, mrow, W1_16, 256, nrow0, 32 * kt, 32 * kt, chunkoff, khalf, acc);
        const _Float16* poolp = s_pool16 + (mtile >> 1) * 128;   // stride-0 broadcast rows
#pragma unroll
        for (int kt = 4; kt < 8; ++kt)
            do_gemm(poolp, 0, mrow, W1_16, 256, nrow0, 32 * kt - 128, 32 * kt, chunkoff, khalf, acc);
        float ones[8] = {1,1,1,1,1,1,1,1};
        epilogue(acc, nt0, lane, mtile, fs + 256, fs + 384, ones, s_h, 128);
    }
    __syncthreads();

    // ---- Stage 5: layer 2, K=128, masked output into s_f (reuse) ----
    {
        v8f acc[4] = {};
#pragma unroll
        for (int kt = 0; kt < 4; ++kt)
            do_gemm(s_h, 128, mrow, W2_16, 128, nrow0, 32 * kt, 32 * kt, chunkoff, khalf, acc);
        float mm[8];
#pragma unroll
        for (int v = 0; v < 8; ++v) mm[v] = s_maskf[mtile * 16 + v + 8 * hi];
        epilogue(acc, nt0, lane, mtile, fs + 512, fs + 640, mm, s_f, 128);
    }
    __syncthreads();

    // ---- Stage 6: second max-pool (f32) + per-polyline validity ----
    {
        int poly = tid >> 7, feat = tid & 127;
        float mx = 0.0f;
#pragma unroll 4
        for (int rr = 0; rr < Nn; ++rr)
            mx = fmaxf(mx, (float)s_f[(poly * Nn + rr) * 128 + feat]);
        s_pool2[poly * 128 + feat] = mx;
        if (tid < 2) {
            float any = 0.0f;
            for (int rr = 0; rr < Nn; ++rr) any = fmaxf(any, s_maskf[tid * Nn + rr]);
            s_valid[tid] = any;
        }
    }
    __syncthreads();

    // ---- Stage 7: head layer 1 (2x128 rows, f32 VALU) ----
    {
        int poly = tid >> 7, o = tid & 127;
        const float* wrow = Wo1 + o * Hn;
        const float* pr = s_pool2 + poly * Hn;
        float sum = bo1[o];
#pragma unroll 4
        for (int k = 0; k < Hn; ++k) sum += pr[k] * wrow[k];
        s_o1[tid] = fmaxf(sum, 0.0f);
    }
    __syncthreads();

    // ---- Stage 8: head layer 2 + validity mask -> global ----
    for (int idx = tid; idx < 2 * On; idx += 256) {
        int poly = idx >> 8, o = idx & 255;
        const float* wrow = Wo2 + o * Hn;
        const float* pr = s_o1 + poly * Hn;
        float sum = bo2[o];
#pragma unroll 4
        for (int k = 0; k < Hn; ++k) sum += pr[k] * wrow[k];
        long gp = rowbase / Nn + poly;       // global polyline index
        out[gp * On + o] = (s_valid[poly] > 0.0f) ? sum : 0.0f;
    }
}

extern "C" void kernel_launch(void* const* d_in, const int* in_sizes, int n_in,
                              void* d_out, int out_size, void* d_ws, size_t ws_size,
                              hipStream_t stream) {
    const float* x      = (const float*)d_in[0];
    const unsigned char* mask = (const unsigned char*)d_in[1];
    const float* W_pre  = (const float*)d_in[2];
    const float* g_pre  = (const float*)d_in[3];
    const float* b_pre  = (const float*)d_in[4];
    const float* m_pre  = (const float*)d_in[5];
    const float* v_pre  = (const float*)d_in[6];
    const float* W1     = (const float*)d_in[7];
    const float* g1     = (const float*)d_in[8];
    const float* b1     = (const float*)d_in[9];
    const float* m1     = (const float*)d_in[10];
    const float* v1     = (const float*)d_in[11];
    const float* W2     = (const float*)d_in[12];
    const float* g2     = (const float*)d_in[13];
    const float* b2     = (const float*)d_in[14];
    const float* m2     = (const float*)d_in[15];
    const float* v2     = (const float*)d_in[16];
    const float* Wo1    = (const float*)d_in[17];
    const float* bo1    = (const float*)d_in[18];
    const float* Wo2    = (const float*)d_in[19];
    const float* bo2    = (const float*)d_in[20];
    float* out = (float*)d_out;

    _Float16* w16 = (_Float16*)d_ws;
    float* fs = (float*)((char*)d_ws + NHALF * sizeof(_Float16));

    // Prolog: f16 weight image + folded BN params (L2-resident thereafter)
    prep_kernel<<<(NHALF + 255) / 256, 256, 0, stream>>>(
        W_pre, W1, W2, g_pre, b_pre, m_pre, v_pre,
        g1, b1, m1, v1, g2, b2, m2, v2, w16, fs);

    // Fused encoder: 2 polylines per workgroup
    int blocks = (Bn * Pn) / 2;              // 6144
    encoder_kernel<<<blocks, 256, 0, stream>>>(x, mask, w16, fs, Wo1, bo1, Wo2, bo2, out);
}
